// QuantizedNN_9895604650610
// MI455X (gfx1250) — compile-verified, matrix operand order checked by reading, not run on hardware
//
#include <hip/hip_runtime.h>

typedef __attribute__((ext_vector_type(16))) __bf16 v16bf;
typedef __attribute__((ext_vector_type(8)))  float  v8f;
typedef __attribute__((ext_vector_type(4)))  float  v4f;

#define THREADS 256
#define IMG_W1   0        /* bf16[4][8][32][16]  A-frags of dequant W1  (32 KB) */
#define IMG_W2   32768    /* bf16[4][2][32][16]  A-frags of int W2       (8 KB) */
#define IMG_B1   40960    /* float[64] */
#define IMG_B2   41216    /* float[64] */
#define IMG_WQ   41472    /* float[64]  a2 * dequant(Wout) */
#define IMG_BYTES 41728
#define CONST_BYTES 64

__device__ __forceinline__ unsigned short f2bf_bits(float f) {
  unsigned u = __float_as_uint(f);
  return (unsigned short)((u + 0x7FFFu + ((u >> 16) & 1u)) >> 16);  // RNE
}
// Per-lane K index for 16-bit A/B fragments (16x32): lanes 0-15 hold K {0..7,16..23},
// lanes 16-31 hold K {8..15,24..31} (ISA 7.12.2 tables).
__device__ __forceinline__ int kmap(int lane, int i) {
  int b = (lane & 16) ? 8 : 0;
  return (i < 8) ? (b + i) : (16 + b + (i - 8));
}
__device__ __forceinline__ v8f wmma_bf16(v16bf a, v16bf b, v8f c) {
  return __builtin_amdgcn_wmma_f32_16x16x32_bf16(false, a, false, b, (short)0, c, false, false);
}

/* ---------------- prep: scales, quantize, pack weight fragments ---------------- */

__device__ float block_maxabs(const float* p, int n, float* red) {
  int t = threadIdx.x;
  float m = 0.f;
  for (int i = t; i < n; i += THREADS) m = fmaxf(m, fabsf(p[i]));
  red[t] = m; __syncthreads();
  for (int s = THREADS / 2; s > 0; s >>= 1) {
    if (t < s) red[t] = fmaxf(red[t], red[t + s]);
    __syncthreads();
  }
  float r = red[0]; __syncthreads();
  return r;
}
__device__ __forceinline__ float p2scale(float maxabs) {
  float m = fmaxf(maxabs, 1e-12f);
  float e = ceilf(log2f(m));
  return exp2f(e) * (1.0f / 128.0f);
}

__global__ void qnn_prep(const float* __restrict__ W1, const float* __restrict__ b1,
                         const float* __restrict__ W2, const float* __restrict__ b2,
                         const float* __restrict__ Wout, const float* __restrict__ bout,
                         const float* __restrict__ a1p, const float* __restrict__ a2p,
                         unsigned char* __restrict__ ws) {
  __shared__ float red[THREADS];
  int t = threadIdx.x;
  float m1 = block_maxabs(W1, 64 * 256, red);
  float m2 = block_maxabs(W2, 64 * 64, red);
  float mo = block_maxabs(Wout, 64, red);
  float s1 = p2scale(m1), s2 = p2scale(m2), so = p2scale(mo);
  float a1 = a1p[0], a2 = a2p[0];
  if (t == 0) {
    float* c = (float*)ws;
    c[0] = 1.0f / a1;   // inv_a1
    c[1] = a1 * s2;     // layer-2 int-accumulator scale
    c[2] = 1.0f / a2;   // inv_a2
    c[3] = bout[0];
  }
  unsigned char* img = ws + CONST_BYTES;
  // W1 dequant values (exact in bf16), packed in WMMA A layout: M=unit, K=feature
  unsigned short* w1p = (unsigned short*)(img + IMG_W1);
  for (int idx = t; idx < 4 * 8 * 32 * 16; idx += THREADS) {
    int i = idx & 15, lane = (idx >> 4) & 31, kc = (idx >> 9) & 7, mt = (idx >> 12) & 3;
    int M = mt * 16 + (lane & 15);
    int K = kc * 32 + kmap(lane, i);
    float w = W1[M * 256 + K];
    float q = fminf(fmaxf(rintf(w / s1), -128.f), 127.f) * s1;
    w1p[idx] = f2bf_bits(q);
  }
  // W2 int8 codes as bf16 (exact), A layout: M=out-unit, K=in-unit
  unsigned short* w2p = (unsigned short*)(img + IMG_W2);
  for (int idx = t; idx < 4 * 2 * 32 * 16; idx += THREADS) {
    int i = idx & 15, lane = (idx >> 4) & 31, kc = (idx >> 9) & 1, mt = (idx >> 10) & 3;
    int M = mt * 16 + (lane & 15);
    int K = kc * 32 + kmap(lane, i);
    float w = W2[M * 64 + K];
    float qi = fminf(fmaxf(rintf(w / s2), -128.f), 127.f);
    w2p[idx] = f2bf_bits(qi);
  }
  float* b1d = (float*)(img + IMG_B1);
  float* b2d = (float*)(img + IMG_B2);
  float* wqd = (float*)(img + IMG_WQ);
  for (int u = t; u < 64; u += THREADS) {
    b1d[u] = b1[u];
    b2d[u] = b2[u];
    float w = Wout[u];
    float q = fminf(fmaxf(rintf(w / so), -128.f), 127.f) * so;
    wqd[u] = a2 * q;   // fold act scale a2 into output weights
  }
}

/* ---------------- fused main kernel: one 16-row tile per wave ---------------- */

__global__ void __launch_bounds__(THREADS)
qnn_fused(const float* __restrict__ x, const unsigned char* __restrict__ ws,
          float* __restrict__ out, int ntiles) {
  __shared__ __align__(64) unsigned char smem[IMG_BYTES];
  int tid = threadIdx.x;
  {
    const unsigned int* src = (const unsigned int*)(ws + CONST_BYTES);
    unsigned int* dst = (unsigned int*)smem;
    for (int i = tid; i < IMG_BYTES / 4; i += THREADS) dst[i] = src[i];
  }
  __syncthreads();

  int lane = tid & 31;
  int wv = tid >> 5;
  int tile = blockIdx.x * 8 + wv;
  if (tile >= ntiles) return;

  const float* cst = (const float*)ws;
  float inv_a1 = cst[0], scale2 = cst[1], inv_a2 = cst[2], outb = cst[3];

  int hi16 = lane >> 4;    // which lane-half (selects K/M sub-offset 8)
  int col = lane & 15;     // batch row within tile (B-operand / C column)
  const float* xrow = x + (size_t)(tile * 16 + col) * 256 + hi16 * 8;

  // ---- layer 1: h1^T = W1q(64x256) @ x^T, bf16 hi/lo split, f32 accum ----
  v8f acc1[4] = {};
  #pragma unroll
  for (int kc = 0; kc < 8; ++kc) {
    const v4f* p0 = (const v4f*)(xrow + kc * 32);
    const v4f* p1 = (const v4f*)(xrow + kc * 32 + 16);
    v4f A0 = __builtin_nontemporal_load(p0);
    v4f A1 = __builtin_nontemporal_load(p0 + 1);
    v4f B0 = __builtin_nontemporal_load(p1);
    v4f B1 = __builtin_nontemporal_load(p1 + 1);
    float f[16] = {A0[0], A0[1], A0[2], A0[3], A1[0], A1[1], A1[2], A1[3],
                   B0[0], B0[1], B0[2], B0[3], B1[0], B1[1], B1[2], B1[3]};
    v16bf bhi, blo;
    #pragma unroll
    for (int i = 0; i < 16; ++i) {
      float v = f[i];
      __bf16 h = (__bf16)v;            // hw RNE cvt (v_cvt_pk_bf16_f32)
      bhi[i] = h;
      blo[i] = (__bf16)(v - (float)h); // exact residual, then hw RNE cvt
    }
    #pragma unroll
    for (int mt = 0; mt < 4; ++mt) {
      const v16bf* a = (const v16bf*)(smem + IMG_W1 + (size_t)((mt * 8 + kc) * 32 + lane) * 32);
      v16bf av = *a;
      acc1[mt] = wmma_bf16(av, bhi, acc1[mt]);
      acc1[mt] = wmma_bf16(av, blo, acc1[mt]);
    }
  }

  // ---- quantize layer-1 output in-lane; C-layout of stacked tiles == B-layout ----
  const float* b1s = (const float*)(smem + IMG_B1) + hi16 * 8;
  v16bf q1c[2];
  #pragma unroll
  for (int mt = 0; mt < 4; ++mt) {
    #pragma unroll
    for (int j = 0; j < 8; ++j) {
      float t = acc1[mt][j] + b1s[mt * 16 + j];
      t = rintf(t * inv_a1);                 // RNE == jnp.round
      t = fminf(fmaxf(t, 0.f), 255.f);       // relu + u8 clip
      q1c[mt >> 1][(mt & 1) * 8 + j] = (__bf16)t;  // ints 0..255 exact in bf16
    }
  }

  // ---- layer 2: exact-integer bf16 WMMA, h2^T = W2int @ q1^T ----
  v8f acc2[4] = {};
  #pragma unroll
  for (int mt = 0; mt < 4; ++mt) {
    #pragma unroll
    for (int k = 0; k < 2; ++k) {
      const v16bf* a = (const v16bf*)(smem + IMG_W2 + (size_t)((mt * 2 + k) * 32 + lane) * 32);
      acc2[mt] = wmma_bf16(*a, q1c[k], acc2[mt]);
    }
  }

  // ---- layer-2 epilogue + layer 3 (64-length dot per batch row) ----
  const float* b2s = (const float*)(smem + IMG_B2) + hi16 * 8;
  const float* wqs = (const float*)(smem + IMG_WQ) + hi16 * 8;
  float part = 0.f;
  #pragma unroll
  for (int mt = 0; mt < 4; ++mt) {
    #pragma unroll
    for (int j = 0; j < 8; ++j) {
      float t = acc2[mt][j] * scale2 + b2s[mt * 16 + j];  // dequant int accum + b2
      t = rintf(t * inv_a2);
      t = fminf(fmaxf(t, 0.f), 255.f);
      part += t * wqs[mt * 16 + j];                        // q2 * (a2 * Woutq)
    }
  }
  // lanes l and l^16 hold complementary halves of the 64 units for the same row
  part += __shfl_xor(part, 16, 32);
  if (hi16 == 0) {
    __builtin_nontemporal_store(part + outb, out + tile * 16 + col);
  }
}

/* ---------------- launch ---------------- */

extern "C" void kernel_launch(void* const* d_in, const int* in_sizes, int n_in,
                              void* d_out, int out_size, void* d_ws, size_t ws_size,
                              hipStream_t stream) {
  const float* x    = (const float*)d_in[0];
  const float* W1   = (const float*)d_in[1];
  const float* b1   = (const float*)d_in[2];
  const float* W2   = (const float*)d_in[3];
  const float* b2   = (const float*)d_in[4];
  const float* Wout = (const float*)d_in[5];
  const float* bout = (const float*)d_in[6];
  const float* a1   = (const float*)d_in[7];
  const float* a2   = (const float*)d_in[8];

  int B = in_sizes[0] / 256;
  int ntiles = B / 16;  // B = 500000 -> 31250 tiles exactly

  qnn_prep<<<1, THREADS, 0, stream>>>(W1, b1, W2, b2, Wout, bout, a1, a2,
                                      (unsigned char*)d_ws);
  int blocks = (ntiles + 7) / 8;
  qnn_fused<<<blocks, THREADS, 0, stream>>>(x, (const unsigned char*)d_ws,
                                            (float*)d_out, ntiles);
}
